// StablePointFlow2DVAE_31499290149580
// MI455X (gfx1250) — matrix operand order repaired and checked by reading, not compiled, not run on hardware
//
#include <hip/hip_runtime.h>

// ---------------------------------------------------------------------------
// StablePointFlow2DVAE cnf_forward — CDNA5 / gfx1250 WMMA implementation (v2).
//
// Key identity: ctx(t) = z + log_det(t)  (div broadcasts over all latent dims),
// so layer 1 is affine in (x0, x1, t, ld) with per-batch precomputed constants.
// Per step: 3 fused GEMMs [65536 x 256 x 256] sharing B=W2 fragments
//   -> one B-load feeds three v_wmma_f32_16x16x32_f16 (h2, tangent0, tangent1).
// ---------------------------------------------------------------------------

typedef __attribute__((ext_vector_type(16))) _Float16 v16h;
typedef __attribute__((ext_vector_type(8)))  float    v8f;

#if defined(__has_builtin)
#if __has_builtin(__builtin_amdgcn_tanhf)
#define FAST_TANH(x) __builtin_amdgcn_tanhf(x)
#endif
#endif
#ifndef FAST_TANH
#define FAST_TANH(x) tanhf(x)
#endif

#define BB     32
#define NN     2048
#define PTS    (BB * NN)     // 65536 points
#define HH     256           // hidden width
#define LATD   128
#define DIND   131
#define NSTEPS 5
#define WPB    2             // waves per block (wave32)

// ---------------------------------------------------------------------------
// Prep kernel 1: W2 (f32 [256][256], row = out-feature n, col = in-feature k)
// -> f16, same layout. B-fragment for wmma is then 16 contiguous halfs/lane.
// ---------------------------------------------------------------------------
__global__ void k_prep_w2(const float* __restrict__ W2, _Float16* __restrict__ W2h) {
    int i = blockIdx.x * blockDim.x + threadIdx.x;
    if (i < HH * HH) W2h[i] = (_Float16)W2[i];
}

// ---------------------------------------------------------------------------
// Prep kernel 2: per-batch layer-1 constants.
//   Wzb[b][f] = b1[f] + sum_j W1[f][3+j] * z[b][j]
//   P1[f]     = { W1[f][0], W1[f][1], W1[f][2], sum_j W1[f][3+j] }
// ---------------------------------------------------------------------------
__global__ void k_prep_wz(const float* __restrict__ W1, const float* __restrict__ b1,
                          const float* __restrict__ z,
                          float* __restrict__ Wzb, float4* __restrict__ P1) {
    int i = blockIdx.x * blockDim.x + threadIdx.x;   // 32*256
    if (i >= BB * HH) return;
    int b = i >> 8, f = i & 255;
    const float* wr = W1 + f * DIND;
    const float* zb = z + b * LATD;
    float acc = b1[f], s = 0.f;
    for (int j = 0; j < LATD; ++j) { float wv = wr[3 + j]; acc += wv * zb[j]; s += wv; }
    Wzb[i] = acc;
    if (b == 0) P1[f] = make_float4(wr[0], wr[1], wr[2], s);
}

// ---------------------------------------------------------------------------
// Main kernel: one wave owns a 16-point tile for all 5 Euler steps.
// ---------------------------------------------------------------------------
__global__ __launch_bounds__(WPB * 32) void k_cnf(
    const float*    __restrict__ x_in,
    const float*    __restrict__ b2g,
    const float*    __restrict__ W3g,     // [2][256]
    const float*    __restrict__ b3g,     // [2]
    const float*    __restrict__ oscg,    // [1]
    const _Float16* __restrict__ W2h,     // [256][256] f16
    const float*    __restrict__ Wzb,     // [32][256]
    const float4*   __restrict__ P1g,     // [256]
    float*          __restrict__ out)     // y[65536][2] then log_det[65536]
{
    __shared__ __attribute__((aligned(32))) float4   sP1[HH];                 // 4 KB
    __shared__ float    sWz[HH];                                              // 1 KB
    __shared__ __attribute__((aligned(32))) _Float16 sDh[WPB][2][8][32][16];  // 32 KB
    __shared__ float    sPt[WPB][16][4];                                      // x0,x1,ld
    __shared__ float    sRed[WPB][32][24];                                    // 6 KB

    const int tid  = threadIdx.x;
    const int w    = tid >> 5;          // wave within block
    const int lane = tid & 31;
    const int tile = blockIdx.x * WPB + w;
    const int pbase = tile * 16;
    const int batch = (blockIdx.x * (WPB * 16)) >> 11;   // same batch for whole block
    const int lh    = lane >> 4;        // half-wave id
    const int nlane = lane & 15;

    for (int f = tid; f < HH; f += WPB * 32) {
        sP1[f] = P1g[f];
        sWz[f] = Wzb[batch * HH + f];
    }
    if (lane < 16) {
        int p = pbase + lane;
        sPt[w][lane][0] = x_in[p * 2 + 0];
        sPt[w][lane][1] = x_in[p * 2 + 1];
        sPt[w][lane][2] = 0.f;
    }
    __syncthreads();

    const float dt = 1.0f / NSTEPS;
    const float oscale = oscg[0];

    v16h h1A[8];   // A-fragments for the h2 GEMM, kept in registers

    for (int st = 0; st < NSTEPS; ++st) {
        const float t  = (float)st * dt;
        const float x0 = sPt[w][nlane][0];
        const float x1 = sPt[w][nlane][1];
        const float ld = sPt[w][nlane][2];

        // ---- layer 1: compute h1 / dh1 fragments directly in WMMA A layout
        #pragma unroll
        for (int kc = 0; kc < 8; ++kc) {
            v16h av;
            #pragma unroll
            for (int s = 0; s < 16; ++s) {
                int koff = s + (lh << 4);   // contiguous k = 16*lh + s (matches B fragment)
                int f = kc * 32 + koff;
                float4 p1 = sP1[f];
                float pre = sWz[f] + x0 * p1.x + x1 * p1.y + t * p1.z + ld * p1.w;
                float h = FAST_TANH(pre);
                float g = 1.0f - h * h;
                av[s] = (_Float16)h;
                sDh[w][0][kc][lane][s] = (_Float16)(g * p1.x);   // g1 * W1[:,0]
                sDh[w][1][kc][lane][s] = (_Float16)(g * p1.y);   // g1 * W1[:,1]
            }
            h1A[kc] = av;
        }
        __syncthreads();

        float vp0[8], vp1[8], dvp[8];
        #pragma unroll
        for (int r = 0; r < 8; ++r) { vp0[r] = 0.f; vp1[r] = 0.f; dvp[r] = 0.f; }

        // ---- fused GEMMs: one B=W2 fragment feeds three WMMAs
        //   accH  = h1  @ W2^T + b2   -> h2 = tanh, g2 = 1-h2^2
        //   accD0 = dh1_0 @ W2^T, accD1 = dh1_1 @ W2^T
        //   v_i  += W3[i]·h2 ; div += g2·(W3[0]·accD0 + W3[1]·accD1)
        for (int nt = 0; nt < 16; ++nt) {
            int n = nt * 16 + nlane;
            float b2n = b2g[n];
            v8f accH, accD0, accD1;
            #pragma unroll
            for (int r = 0; r < 8; ++r) { accH[r] = b2n; accD0[r] = 0.f; accD1[r] = 0.f; }
            const _Float16* bp = W2h + n * HH + lh * 16;
            #pragma unroll
            for (int kc = 0; kc < 8; ++kc) {
                v16h bfrag = *(const v16h*)(bp + kc * 32);
                accH  = __builtin_amdgcn_wmma_f32_16x16x32_f16(
                          false, h1A[kc], false, bfrag, (short)0, accH, false, false);
                v16h a0 = *(const v16h*)(&sDh[w][0][kc][lane][0]);
                accD0 = __builtin_amdgcn_wmma_f32_16x16x32_f16(
                          false, a0, false, bfrag, (short)0, accD0, false, false);
                v16h a1 = *(const v16h*)(&sDh[w][1][kc][lane][0]);
                accD1 = __builtin_amdgcn_wmma_f32_16x16x32_f16(
                          false, a1, false, bfrag, (short)0, accD1, false, false);
            }
            float w30 = W3g[n], w31 = W3g[HH + n];
            #pragma unroll
            for (int r = 0; r < 8; ++r) {     // element r: M = r + 8*lh, N = n
                float h2 = FAST_TANH(accH[r]);
                float g2 = 1.0f - h2 * h2;
                vp0[r] += w30 * h2;
                vp1[r] += w31 * h2;
                dvp[r] += g2 * (w30 * accD0[r] + w31 * accD1[r]);
            }
        }

        // ---- cross-lane reduction of v / div partials, Euler update
        #pragma unroll
        for (int r = 0; r < 8; ++r) {
            sRed[w][lane][r]      = vp0[r];
            sRed[w][lane][8 + r]  = vp1[r];
            sRed[w][lane][16 + r] = dvp[r];
        }
        __syncthreads();
        if (lane < 16) {
            int m = lane;
            int lb = (m >> 3) << 4;   // lanes 0..15 hold M 0..7; lanes 16..31 hold M 8..15
            int r  = m & 7;
            float s0 = 0.f, s1 = 0.f, sd = 0.f;
            #pragma unroll
            for (int ll = 0; ll < 16; ++ll) {
                s0 += sRed[w][lb + ll][r];
                s1 += sRed[w][lb + ll][8 + r];
                sd += sRed[w][lb + ll][16 + r];
            }
            float vx = (s0 + b3g[0]) * oscale;
            float vy = (s1 + b3g[1]) * oscale;
            float dd = sd * oscale;
            sPt[w][m][0] += vx * dt;
            sPt[w][m][1] += vy * dt;
            sPt[w][m][2] += dd * dt;
        }
        __syncthreads();
    }

    if (lane < 16) {
        int p = pbase + lane;
        out[p * 2 + 0]   = sPt[w][lane][0];
        out[p * 2 + 1]   = sPt[w][lane][1];
        out[2 * PTS + p] = sPt[w][lane][2];
    }
}

// ---------------------------------------------------------------------------
extern "C" void kernel_launch(void* const* d_in, const int* in_sizes, int n_in,
                              void* d_out, int out_size, void* d_ws, size_t ws_size,
                              hipStream_t stream) {
    (void)in_sizes; (void)n_in; (void)out_size; (void)ws_size;

    const float* x   = (const float*)d_in[0];
    const float* z   = (const float*)d_in[1];
    const float* W1  = (const float*)d_in[2];
    const float* b1  = (const float*)d_in[3];
    const float* W2  = (const float*)d_in[4];
    const float* b2  = (const float*)d_in[5];
    const float* W3  = (const float*)d_in[6];
    const float* b3  = (const float*)d_in[7];
    const float* osc = (const float*)d_in[8];
    float* out = (float*)d_out;

    // workspace layout
    _Float16* W2h = (_Float16*)d_ws;                                   // 131072 B
    float*    Wzb = (float*)((char*)d_ws + 131072);                    //  32768 B
    float4*   P1  = (float4*)((char*)d_ws + 131072 + 32768);           //   4096 B

    k_prep_w2<<<(HH * HH) / 256, 256, 0, stream>>>(W2, W2h);
    k_prep_wz<<<(BB * HH) / 256, 256, 0, stream>>>(W1, b1, z, Wzb, P1);

    dim3 grid(PTS / (16 * WPB));          // 2048 blocks
    dim3 block(WPB * 32);                 // 64 threads = 2 waves
    k_cnf<<<grid, block, 0, stream>>>(x, b2, W3, b3, osc, W2h, Wzb, P1, out);
}
